// QueryGuidedFusionNet_4166118277705
// MI455X (gfx1250) — matrix operand order repaired and checked by reading, not compiled, not run on hardware
//
#include <hip/hip_runtime.h>
#include <hip/hip_bf16.h>
#include <stdint.h>

// ---------------------------------------------------------------------------
// CDNA5 (gfx1250) WMMA-based implementation of QueryGuidedFusionNet forward.
// All matmuls use v_wmma_f32_16x16x32_bf16 (fp32 -> bf16 RNE, fp32 accum).
// ---------------------------------------------------------------------------

typedef __attribute__((ext_vector_type(16))) __bf16 v16bf;
typedef __attribute__((ext_vector_type(8)))  float  v8f;

union FragBF { v16bf v; uint32_t u[8]; };

// Native fp32->bf16 conversion (RNE); lowers to v_cvt_*bf16* on gfx1250,
// far cheaper than a manual integer rounding sequence.
__device__ __forceinline__ uint32_t pack2_bf16(float x, float y) {
  union { uint32_t u; __bf16 h[2]; } p;
  p.h[0] = (__bf16)x;
  p.h[1] = (__bf16)y;
  return p.u;
}

__device__ __forceinline__ float gelu_exact(float x) {
  return 0.5f * x * (1.0f + erff(x * 0.70710678118654752f));
}

__device__ __forceinline__ v8f wmma_bf16(const FragBF& a, const FragBF& b, v8f c) {
  return __builtin_amdgcn_wmma_f32_16x16x32_bf16(false, a.v, false, b.v,
                                                 (short)0, c, false, false);
}

// ---------------------------------------------------------------------------
// Generic batched GEMM:  C = scale * (A @ B(^T)) [+ bias] [GELU] [+ resid]
//   B_IS_NT = true : Bm is W[N,K] row-major (C = A · W^T)
//   B_IS_NT = false: Bm is B[K,N] row-major (C = A · B)
//   ALIGNED = true : M%64==0, N%64==0, K%32==0, rows 16B-aligned -> no guards,
//                    float4 (b128) global loads, b64 LDS stores.
// Block tile 64x64, BK=32, 128 threads = 4 waves; wave (mi,ni) owns a 32x32
// patch = 2 A-frags x 2 B-frags -> 4 WMMAs/k-step with 2x fragment reuse.
// LDS tiles hold bf16 pairs pre-arranged in WMMA operand order.
// ---------------------------------------------------------------------------
template<bool B_IS_NT, bool ALIGNED, bool DO_GELU>
__global__ __launch_bounds__(128)
void wmma_gemm_kernel(const float* __restrict__ A,
                      const float* __restrict__ Bm,
                      const float* __restrict__ bias,
                      const float* __restrict__ resid,
                      float* __restrict__ C,
                      int M, int N, int K,
                      int lda, int ldb, int ldc,
                      long sAb, long sAh, long sBb, long sBh,
                      long sCb, long sCh,
                      int H, float scale)
{
  __shared__ uint32_t As[64][17];   // [row][kpair], +1 pad vs bank conflicts
  __shared__ uint32_t Bs[64][17];   // [col][kpair]

  const int z  = blockIdx.z;
  const int zb = z / H;
  const int zh = z - zb * H;
  A  += (long)zb * sAb + (long)zh * sAh;
  Bm += (long)zb * sBb + (long)zh * sBh;
  const long cOff = (long)zb * sCb + (long)zh * sCh;

  const int tid  = threadIdx.x;
  const int lane = tid & 31;
  const int wave = tid >> 5;
  const int mi   = wave >> 1;       // 0..1 : 32-row group
  const int ni   = wave & 1;        // 0..1 : 32-col group
  const int half = lane >> 4;       // 0: lanes 0-15, 1: lanes 16-31
  const int lr   = lane & 15;

  const int blockM = blockIdx.x * 64;
  const int blockN = blockIdx.y * 64;

  v8f acc00 = {}, acc01 = {}, acc10 = {}, acc11 = {};

  for (int k0 = 0; k0 < K; k0 += 32) {
    if (ALIGNED) {
      // ---- A tile: 64 rows x 32 k -> 512 float4 loads, 128 thr x 4 ----
#pragma unroll
      for (int i = 0; i < 4; ++i) {
        int idx = tid + i * 128;           // 0..511
        int r = idx >> 3, q4 = idx & 7;
        const float4 a = *(const float4*)(A + (long)(blockM + r) * lda + k0 + 4 * q4);
        As[r][2 * q4]     = pack2_bf16(a.x, a.y);
        As[r][2 * q4 + 1] = pack2_bf16(a.z, a.w);
      }
      if (B_IS_NT) {
#pragma unroll
        for (int i = 0; i < 4; ++i) {
          int idx = tid + i * 128;
          int n = idx >> 3, q4 = idx & 7;
          const float4 b = *(const float4*)(Bm + (long)(blockN + n) * ldb + k0 + 4 * q4);
          Bs[n][2 * q4]     = pack2_bf16(b.x, b.y);
          Bs[n][2 * q4 + 1] = pack2_bf16(b.z, b.w);
        }
      } else {
        // B[k][n]: k-strided pair loads (no guards)
#pragma unroll
        for (int i = 0; i < 8; ++i) {
          int idx = tid + i * 128;         // 0..1023
          int n = idx >> 4, p = idx & 15;
          const float* bp = Bm + (long)(k0 + 2 * p) * ldb + blockN + n;
          Bs[n][p] = pack2_bf16(bp[0], bp[ldb]);
        }
      }
    } else {
      // ---- guarded path (used only for LR=196-shaped GEMMs) ----
#pragma unroll
      for (int i = 0; i < 8; ++i) {
        int idx = tid + i * 128;
        int r = idx >> 4, p = idx & 15;
        int gr = blockM + r, gk = k0 + 2 * p;
        float a0 = 0.f, a1 = 0.f;
        if (gr < M && gk < K) {
          const float* ap = A + (long)gr * lda + gk;
          a0 = ap[0];
          if (gk + 1 < K) a1 = ap[1];
        }
        As[r][p] = pack2_bf16(a0, a1);
      }
#pragma unroll
      for (int i = 0; i < 8; ++i) {
        int idx = tid + i * 128;
        int n = idx >> 4, p = idx & 15;
        int gn = blockN + n, gk = k0 + 2 * p;
        float b0 = 0.f, b1 = 0.f;
        if (gn < N) {
          if (B_IS_NT) {
            if (gk < K) {
              const float* bp = Bm + (long)gn * ldb + gk;
              b0 = bp[0];
              if (gk + 1 < K) b1 = bp[1];
            }
          } else {
            if (gk < K)     b0 = Bm[(long)gk * ldb + gn];
            if (gk + 1 < K) b1 = Bm[(long)(gk + 1) * ldb + gn];
          }
        }
        Bs[n][p] = pack2_bf16(b0, b1);
      }
    }
    __syncthreads();

    if (ALIGNED && (k0 + 32 < K)) {  // speculative prefetch of next A tile
      __builtin_prefetch(A + (long)(blockM + (tid >> 1)) * lda + (k0 + 32), 0, 1);
    }

    // B fragment (32x16, 16-bit): VGPR v <- K pair (half*8 + v)
    // A fragment (16x32, 16-bit): VGPR v <- K pair (half*8 + v)
    FragBF fb0, fb1, fa0, fa1;
#pragma unroll
    for (int v = 0; v < 8; ++v) {
      const int pB = (half << 3) + v;
      fb0.u[v] = Bs[ni * 32 + lr][pB];
      fb1.u[v] = Bs[ni * 32 + 16 + lr][pB];
      const int pA = (half << 3) + v;
      fa0.u[v] = As[mi * 32 + lr][pA];
      fa1.u[v] = As[mi * 32 + 16 + lr][pA];
    }

    acc00 = wmma_bf16(fa0, fb0, acc00);
    acc01 = wmma_bf16(fa0, fb1, acc01);
    acc10 = wmma_bf16(fa1, fb0, acc10);
    acc11 = wmma_bf16(fa1, fb1, acc11);
    __syncthreads();
  }

  // ---- epilogue: C/D layout: VGPR r -> row r + half*8, col = lane&15 ----
  C += cOff;
  const float* R = resid ? resid + cOff : nullptr;

  v8f* accs[2][2] = { { &acc00, &acc01 }, { &acc10, &acc11 } };
#pragma unroll
  for (int nt = 0; nt < 2; ++nt) {
    const int col = blockN + ni * 32 + nt * 16 + lr;
    if (ALIGNED || col < N) {
      const float bv = bias ? bias[col] : 0.f;
#pragma unroll
      for (int mt = 0; mt < 2; ++mt) {
        const v8f& a = *accs[mt][nt];
#pragma unroll
        for (int r = 0; r < 8; ++r) {
          const int row = blockM + mi * 32 + mt * 16 + r + (half << 3);
          if (ALIGNED || row < M) {
            float v = a[r] * scale + bv;
            if (DO_GELU) v = gelu_exact(v);
            if (R) v += R[(long)row * ldc + col];
            C[(long)row * ldc + col] = v;
          }
        }
      }
    }
  }
}

// ---------------------------------------------------------------------------
// Masked softmax over last dim (scale already folded into the scores GEMM).
// grid = (NQ, batch); one block (256 thr, 8 wave32) per score row.
// ---------------------------------------------------------------------------
__global__ __launch_bounds__(256)
void softmax_mask_kernel(float* __restrict__ S, const int* __restrict__ mask,
                         int Lk, int NQ, int H)
{
  const int z  = blockIdx.y;
  const int zb = z / H;
  float* row = S + ((long)z * NQ + blockIdx.x) * Lk;
  const int* m = mask + (long)zb * Lk;
  const int tid = threadIdx.x;

  __shared__ float sbuf[8];
  __shared__ float sval;

  float mx = -3.0e38f;
  for (int i = tid; i < Lk; i += 256)
    if (m[i] != 0) mx = fmaxf(mx, row[i]);
#pragma unroll
  for (int off = 16; off > 0; off >>= 1) mx = fmaxf(mx, __shfl_down(mx, off, 32));
  if ((tid & 31) == 0) sbuf[tid >> 5] = mx;
  __syncthreads();
  if (tid == 0) {
    float v = sbuf[0];
    for (int i = 1; i < 8; ++i) v = fmaxf(v, sbuf[i]);
    sval = v;
  }
  __syncthreads();
  mx = sval;

  float sum = 0.f;
  for (int i = tid; i < Lk; i += 256) {
    float v = (m[i] != 0) ? __expf(row[i] - mx) : 0.f;
    row[i] = v;
    sum += v;
  }
  __syncthreads();
#pragma unroll
  for (int off = 16; off > 0; off >>= 1) sum += __shfl_down(sum, off, 32);
  if ((tid & 31) == 0) sbuf[tid >> 5] = sum;
  __syncthreads();
  if (tid == 0) {
    float v = 0.f;
    for (int i = 0; i < 8; ++i) v += sbuf[i];
    sval = 1.f / v;
  }
  __syncthreads();
  const float inv = sval;
  for (int i = tid; i < Lk; i += 256) row[i] *= inv;
}

__device__ __forceinline__ float block_sum_256(float v, float* sbuf, float* sval, int tid) {
#pragma unroll
  for (int off = 16; off > 0; off >>= 1) v += __shfl_down(v, off, 32);
  if ((tid & 31) == 0) sbuf[tid >> 5] = v;
  __syncthreads();
  if (tid == 0) {
    float s = 0.f;
    for (int i = 0; i < 8; ++i) s += sbuf[i];
    *sval = s;
  }
  __syncthreads();
  float r = *sval;
  __syncthreads();
  return r;
}

// O = LayerNorm(X + Y) * g + b ; one block per row, D = 1024.
__global__ __launch_bounds__(256)
void add_ln_kernel(const float* __restrict__ X, const float* __restrict__ Y,
                   const float* __restrict__ g, const float* __restrict__ b,
                   float* __restrict__ O, int Dm)
{
  const long row = blockIdx.x;
  const float* x = X + row * Dm;
  const float* y = Y + row * Dm;
  float* o = O + row * Dm;
  const int tid = threadIdx.x;
  __shared__ float sbuf[8];
  __shared__ float sval;

  float vals[4];
  float s = 0.f;
#pragma unroll
  for (int i = 0; i < 4; ++i) {
    int idx = tid + i * 256;
    vals[i] = x[idx] + y[idx];
    s += vals[i];
  }
  const float mu = block_sum_256(s, sbuf, &sval, tid) / (float)Dm;

  float s2 = 0.f;
#pragma unroll
  for (int i = 0; i < 4; ++i) {
    float d = vals[i] - mu;
    s2 += d * d;
  }
  const float var  = block_sum_256(s2, sbuf, &sval, tid) / (float)Dm;
  const float rstd = rsqrtf(var + 1e-5f);
#pragma unroll
  for (int i = 0; i < 4; ++i) {
    int idx = tid + i * 256;
    o[idx] = (vals[i] - mu) * rstd * g[idx] + b[idx];
  }
}

// out = q + alpha*aggr + (1-alpha)*aggt
__global__ void combine_kernel(const float* __restrict__ q,
                               const float* __restrict__ aggr,
                               const float* __restrict__ aggt,
                               const float* __restrict__ alpha,
                               float* __restrict__ out, int n)
{
  int i = blockIdx.x * blockDim.x + threadIdx.x;
  if (i < n) {
    float a = alpha[0];
    out[i] = q[i] + a * aggr[i] + (1.f - a) * aggt[i];
  }
}

// ---------------------------------------------------------------------------
extern "C" void kernel_launch(void* const* d_in, const int* in_sizes, int n_in,
                              void* d_out, int out_size, void* d_ws, size_t ws_size,
                              hipStream_t stream)
{
  (void)in_sizes; (void)n_in; (void)out_size; (void)ws_size;

  constexpr int Bn = 16, NQ = 128, LT = 512, LR = 196, Dm = 1024, H = 16, dh = 64, F = 4096;

  const float* queries = (const float*)d_in[0];
  const float* text    = (const float*)d_in[1];
  const float* region  = (const float*)d_in[2];
  const int*   mask_t  = (const int*)d_in[3];
  const int*   mask_r  = (const int*)d_in[4];
  const float* w_qkv_t = (const float*)d_in[5];
  const float* b_qkv_t = (const float*)d_in[6];
  const float* w_o_t   = (const float*)d_in[7];
  const float* b_o_t   = (const float*)d_in[8];
  const float* w_qkv_r = (const float*)d_in[9];
  const float* b_qkv_r = (const float*)d_in[10];
  const float* w_o_r   = (const float*)d_in[11];
  const float* b_o_r   = (const float*)d_in[12];
  const float* ln_g    = (const float*)d_in[13];
  const float* ln_b    = (const float*)d_in[14];
  const float* w1      = (const float*)d_in[15];
  const float* b1      = (const float*)d_in[16];
  const float* w2      = (const float*)d_in[17];
  const float* b2      = (const float*)d_in[18];
  const float* alpha   = (const float*)d_in[19];
  float* outp = (float*)d_out;

  // Workspace layout (floats), with buffer reuse; total = 44,040,192 f = 168 MiB.
  float* ws   = (float*)d_ws;
  float* q    = ws + 0L;          //  2,097,152  Q projection (both MHAs)
  float* kt   = ws + 2097152L;    //  8,388,608  text K ; later region K
  float* vr_  = ws + 5308416L;    //             region V (aliases kt tail)
  float* vt   = ws + 10485760L;   //  8,388,608  text V ; later agg_t
  float* aggt = vt;
  float* aggr = ws + 12582912L;
  float* qc   = ws + 14680064L;
  float* sc   = ws + 18874368L;   // 16,777,216  scores ; later FFN hidden
  float* ao   = ws + 35651584L;   //  2,097,152  attention output (pre o-proj)
  float* mo   = ws + 37748736L;   //  2,097,152  o-proj output
  float* q1   = ws + 39845888L;   //  2,097,152  after first residual+LN
  float* q2   = ws + 41943040L;   //  2,097,152  after second residual+LN
  float* kr_  = kt;

  const int MBQ = Bn * NQ;   // 2048
  const int MBT = Bn * LT;   // 8192
  const int MBR = Bn * LR;   // 3136 (multiple of 64)

  auto cdiv = [](int a, int b) { return (a + b - 1) / b; };
  auto grid = [&](int M_, int N_, int Z_) {
    return dim3((unsigned)cdiv(M_, 64), (unsigned)cdiv(N_, 64), (unsigned)Z_);
  };
  const float inv_sqrt_dh = 0.125f;     // 1/sqrt(64)
  const float inv_sqrt_D  = 0.03125f;   // 1/sqrt(1024)

  // ---------------- text MHA ----------------
  wmma_gemm_kernel<true,true,false><<<grid(MBQ, Dm, 1), 128, 0, stream>>>(
      queries, w_qkv_t, b_qkv_t, nullptr, q,
      MBQ, Dm, Dm, Dm, Dm, Dm, 0,0,0,0,0,0, 1, 1.f);
  wmma_gemm_kernel<true,true,false><<<grid(MBT, Dm, 1), 128, 0, stream>>>(
      text, w_qkv_t + (long)Dm*Dm, b_qkv_t + Dm, nullptr, kt,
      MBT, Dm, Dm, Dm, Dm, Dm, 0,0,0,0,0,0, 1, 1.f);
  wmma_gemm_kernel<true,true,false><<<grid(MBT, Dm, 1), 128, 0, stream>>>(
      text, w_qkv_t + 2L*Dm*Dm, b_qkv_t + 2*Dm, nullptr, vt,
      MBT, Dm, Dm, Dm, Dm, Dm, 0,0,0,0,0,0, 1, 1.f);
  wmma_gemm_kernel<true,true,false><<<grid(NQ, LT, Bn*H), 128, 0, stream>>>(
      q, kt, nullptr, nullptr, sc,
      NQ, LT, dh, Dm, Dm, LT,
      (long)NQ*Dm, dh, (long)LT*Dm, dh, (long)H*NQ*LT, (long)NQ*LT, H, inv_sqrt_dh);
  softmax_mask_kernel<<<dim3(NQ, Bn*H), 256, 0, stream>>>(sc, mask_t, LT, NQ, H);
  wmma_gemm_kernel<false,true,false><<<grid(NQ, dh, Bn*H), 128, 0, stream>>>(
      sc, vt, nullptr, nullptr, ao,
      NQ, dh, LT, LT, Dm, Dm,
      (long)H*NQ*LT, (long)NQ*LT, (long)LT*Dm, dh, (long)NQ*Dm, dh, H, 1.f);
  wmma_gemm_kernel<true,true,false><<<grid(MBQ, Dm, 1), 128, 0, stream>>>(
      ao, w_o_t, b_o_t, nullptr, mo,
      MBQ, Dm, Dm, Dm, Dm, Dm, 0,0,0,0,0,0, 1, 1.f);
  add_ln_kernel<<<dim3(MBQ), 256, 0, stream>>>(queries, mo, ln_g, ln_b, q1, Dm);

  // ---------------- region MHA ----------------
  wmma_gemm_kernel<true,true,false><<<grid(MBQ, Dm, 1), 128, 0, stream>>>(
      q1, w_qkv_r, b_qkv_r, nullptr, q,
      MBQ, Dm, Dm, Dm, Dm, Dm, 0,0,0,0,0,0, 1, 1.f);
  wmma_gemm_kernel<true,true,false><<<grid(MBR, Dm, 1), 128, 0, stream>>>(
      region, w_qkv_r + (long)Dm*Dm, b_qkv_r + Dm, nullptr, kr_,
      MBR, Dm, Dm, Dm, Dm, Dm, 0,0,0,0,0,0, 1, 1.f);
  wmma_gemm_kernel<true,true,false><<<grid(MBR, Dm, 1), 128, 0, stream>>>(
      region, w_qkv_r + 2L*Dm*Dm, b_qkv_r + 2*Dm, nullptr, vr_,
      MBR, Dm, Dm, Dm, Dm, Dm, 0,0,0,0,0,0, 1, 1.f);
  wmma_gemm_kernel<true,false,false><<<grid(NQ, LR, Bn*H), 128, 0, stream>>>(
      q, kr_, nullptr, nullptr, sc,
      NQ, LR, dh, Dm, Dm, LR,
      (long)NQ*Dm, dh, (long)LR*Dm, dh, (long)H*NQ*LR, (long)NQ*LR, H, inv_sqrt_dh);
  softmax_mask_kernel<<<dim3(NQ, Bn*H), 256, 0, stream>>>(sc, mask_r, LR, NQ, H);
  wmma_gemm_kernel<false,false,false><<<grid(NQ, dh, Bn*H), 128, 0, stream>>>(
      sc, vr_, nullptr, nullptr, ao,
      NQ, dh, LR, LR, Dm, Dm,
      (long)H*NQ*LR, (long)NQ*LR, (long)LR*Dm, dh, (long)NQ*Dm, dh, H, 1.f);
  wmma_gemm_kernel<true,true,false><<<grid(MBQ, Dm, 1), 128, 0, stream>>>(
      ao, w_o_r, b_o_r, nullptr, mo,
      MBQ, Dm, Dm, Dm, Dm, Dm, 0,0,0,0,0,0, 1, 1.f);
  add_ln_kernel<<<dim3(MBQ), 256, 0, stream>>>(q1, mo, ln_g, ln_b, q2, Dm);

  // ---------------- manual masked cross-attentions (scale 1/sqrt(D)) --------
  wmma_gemm_kernel<true,true,false><<<grid(NQ, LT, Bn), 128, 0, stream>>>(
      q2, text, nullptr, nullptr, sc,
      NQ, LT, Dm, Dm, Dm, LT,
      (long)NQ*Dm, 0, (long)LT*Dm, 0, (long)NQ*LT, 0, 1, inv_sqrt_D);
  softmax_mask_kernel<<<dim3(NQ, Bn), 256, 0, stream>>>(sc, mask_t, LT, NQ, 1);
  wmma_gemm_kernel<false,true,false><<<grid(NQ, Dm, Bn), 128, 0, stream>>>(
      sc, text, nullptr, nullptr, aggt,
      NQ, Dm, LT, LT, Dm, Dm,
      (long)NQ*LT, 0, (long)LT*Dm, 0, (long)NQ*Dm, 0, 1, 1.f);

  wmma_gemm_kernel<true,false,false><<<grid(NQ, LR, Bn), 128, 0, stream>>>(
      q2, region, nullptr, nullptr, sc,
      NQ, LR, Dm, Dm, Dm, LR,
      (long)NQ*Dm, 0, (long)LR*Dm, 0, (long)NQ*LR, 0, 1, inv_sqrt_D);
  softmax_mask_kernel<<<dim3(NQ, Bn), 256, 0, stream>>>(sc, mask_r, LR, NQ, 1);
  wmma_gemm_kernel<false,false,false><<<grid(NQ, Dm, Bn), 128, 0, stream>>>(
      sc, region, nullptr, nullptr, aggr,
      NQ, Dm, LR, LR, Dm, Dm,
      (long)NQ*LR, 0, (long)LR*Dm, 0, (long)NQ*Dm, 0, 1, 1.f);

  const int nElem = MBQ * Dm;
  combine_kernel<<<dim3(cdiv(nElem, 256)), 256, 0, stream>>>(q2, aggr, aggt, alpha, qc, nElem);

  // ---------------- FFN: qc + W2·GELU(W1·qc + b1) + b2 -----------------------
  wmma_gemm_kernel<true,true,true><<<grid(MBQ, F, 1), 128, 0, stream>>>(
      qc, w1, b1, nullptr, sc,
      MBQ, F, Dm, Dm, Dm, F, 0,0,0,0,0,0, 1, 1.f);
  wmma_gemm_kernel<true,true,false><<<grid(MBQ, Dm, 1), 128, 0, stream>>>(
      sc, w2, b2, qc, outp,
      MBQ, Dm, F, F, F, Dm, 0,0,0,0,0,0, 1, 1.f);
}
